// GenGPT_Edge_78864189489802
// MI455X (gfx1250) — compile-verified, matrix operand order checked by reading, not run on hardware
//
#include <hip/hip_runtime.h>
#include <stdint.h>

// ---------------------------------------------------------------------------
// Problem constants (from reference)
// ---------------------------------------------------------------------------
#define BB   32
#define TT   128
#define BT   (BB*TT)          // 4096
#define DD   512
#define HH   8
#define DH   64
#define DFF  2048
#define LL   6
#define NE_MAX (BB * (TT-1) * TT / 2)   // 260096 max edges

// ---------------------------------------------------------------------------
// CDNA5 WMMA types / helpers (wave32, v_wmma_f32_16x16x32_bf16)
// ---------------------------------------------------------------------------
typedef __attribute__((ext_vector_type(16))) __bf16 v16bf;
typedef __attribute__((ext_vector_type(8)))  float  v8f;

union FragU { v16bf v; uint32_t u[8]; };

__device__ __forceinline__ uint16_t f2bf(float f) {
    union { float f; uint32_t u; } c; c.f = f;
    uint32_t u = c.u;
    uint32_t r = u + 0x7FFFu + ((u >> 16) & 1u);   // round-to-nearest-even
    return (uint16_t)(r >> 16);
}

// Gather a 16x32 bf16 A/B fragment from LDS laid out row-major
// (row = M for A, row = N for B-transposed), per ISA 7.12.2:
//   lanes 0-15: row=lane, K base 0 ; lanes 16-31: row=lane-16, K base 8
//   pair p<4 -> K = base+2p ; p>=4 -> K = base+16+2(p-4)
__device__ __forceinline__ v16bf frag16(const uint16_t* base, int stride, int lane, int k0) {
    const int m  = lane & 15;
    const int kb = k0 + ((lane & 16) ? 8 : 0);
    FragU f;
#pragma unroll
    for (int p = 0; p < 8; ++p) {
        const int kk = kb + ((p < 4) ? (2*p) : (16 + 2*(p-4)));
        f.u[p] = *reinterpret_cast<const uint32_t*>(base + m * stride + kk);
    }
    return f.v;
}

__device__ __forceinline__ v8f wmma_bf16(v16bf a, v16bf b, v8f c) {
    return __builtin_amdgcn_wmma_f32_16x16x32_bf16(false, a, false, b, (short)0, c, false, false);
}

// CDNA5 async global->LDS copy (ASYNCcnt-tracked, ISA 08_async_tensor §4).
// 16 bytes per lane land directly in LDS; generic shared pointer's low 32
// bits are the LDS address.
__device__ __forceinline__ void async_load_b128(void* lds_ptr, const void* gaddr) {
    uint32_t lds_addr = (uint32_t)(uintptr_t)lds_ptr;
    asm volatile("global_load_async_to_lds_b128 %0, %1, off"
                 :: "v"(lds_addr), "v"(gaddr)
                 : "memory");
}
__device__ __forceinline__ void wait_asynccnt0() {
    asm volatile("s_wait_asynccnt 0x0" ::: "memory");
}

__device__ __forceinline__ float xorred_add16(float v) {
    v += __shfl_xor(v, 1, 32); v += __shfl_xor(v, 2, 32);
    v += __shfl_xor(v, 4, 32); v += __shfl_xor(v, 8, 32);
    return v;
}
__device__ __forceinline__ float xorred_max16(float v) {
    v = fmaxf(v, __shfl_xor(v, 1, 32)); v = fmaxf(v, __shfl_xor(v, 2, 32));
    v = fmaxf(v, __shfl_xor(v, 4, 32)); v = fmaxf(v, __shfl_xor(v, 8, 32));
    return v;
}
__device__ __forceinline__ float gelu_exact(float x) {
    return 0.5f * x * (1.0f + erff(x * 0.70710678118654752f));
}

// ---------------------------------------------------------------------------
// Small utility kernels
// ---------------------------------------------------------------------------
__global__ void convert_bf16_kernel(const float* __restrict__ s, uint16_t* __restrict__ d, int n) {
    int i = blockIdx.x * blockDim.x + threadIdx.x;
    if (i < n) d[i] = f2bf(s[i]);
}

__global__ void embed_kernel(const int* __restrict__ tokens, const float* __restrict__ emb,
                             float* __restrict__ X) {
    const int r = blockIdx.x;                 // 0..BT-1
    const int tok = tokens[r * 2];
    for (int i = threadIdx.x; i < DD; i += blockDim.x)
        X[r * DD + i] = emb[tok * DD + i];
}

// ctl[0]=NN ctl[1]=NE ctl[2..33]=node offsets ctl[34..65]=edge offsets
__global__ void prep_kernel(const int* __restrict__ nn, int* __restrict__ ctl) {
    if (threadIdx.x == 0 && blockIdx.x == 0) {
        int noff = 0, eoff = 0;
        for (int b = 0; b < BB; ++b) {
            ctl[2 + b]  = noff;
            ctl[34 + b] = eoff;
            const int n = nn[b];
            noff += n;
            eoff += n * (n - 1) / 2;
        }
        ctl[0] = noff;
        ctl[1] = eoff;
    }
}

__global__ void rowmap_kernel(const int* __restrict__ nn, const int* __restrict__ ctl,
                              int* __restrict__ row_map) {
    const int b  = blockIdx.x / (TT - 1);
    const int pi = blockIdx.x % (TT - 1);
    const int n  = nn[b];
    if (pi >= n - 1) return;
    const int base = ctl[34 + b] + pi * (pi + 1) / 2;
    for (int pj = threadIdx.x; pj <= pi; pj += blockDim.x)
        row_map[base + pj] = (b << 16) | (pi << 8) | pj;
}

// LayerNorm (D=512) -> bf16. One block (256 thr) per row, 2 elems/thread.
__global__ __launch_bounds__(256) void ln_bf16_kernel(const float* __restrict__ X,
                                                      const float* __restrict__ W,
                                                      uint16_t* __restrict__ Out) {
    __shared__ float red[8];
    const int r = blockIdx.x, tid = threadIdx.x;
    const int lane = tid & 31, wave = tid >> 5;
    const float a = X[r * DD + tid];
    const float b = X[r * DD + 256 + tid];
    float s = a + b;
#pragma unroll
    for (int m = 16; m >= 1; m >>= 1) s += __shfl_xor(s, m, 32);
    if (lane == 0) red[wave] = s;
    __syncthreads();
    float mu = 0.0f;
#pragma unroll
    for (int i = 0; i < 8; ++i) mu += red[i];
    mu *= (1.0f / (float)DD);
    const float da = a - mu, db = b - mu;
    float v = da * da + db * db;
#pragma unroll
    for (int m = 16; m >= 1; m >>= 1) v += __shfl_xor(v, m, 32);
    __syncthreads();                 // done reading red before overwrite
    if (lane == 0) red[wave] = v;
    __syncthreads();
    float var = 0.0f;
#pragma unroll
    for (int i = 0; i < 8; ++i) var += red[i];
    var *= (1.0f / (float)DD);
    const float rstd = rsqrtf(var + 1e-5f);
    Out[r * DD + tid]        = f2bf(da * rstd * W[tid]);
    Out[r * DD + 256 + tid]  = f2bf(db * rstd * W[256 + tid]);
}

// ---------------------------------------------------------------------------
// Tiled bf16 WMMA GEMM: C[M,N] = epilogue(A[M,K]bf16 @ B[K,N]bf16)
// Block: 256 thr (8 waves), C tile 128x128; wave w -> rows w*16..w*16+15,
// all 128 cols (8 WMMA accumulators). A staged via async global->LDS b128;
// B staged transposed ([n][k]) so fragment gathers are 8x ds_load_b32.
// ---------------------------------------------------------------------------
enum { EP_F32 = 0, EP_RES_F32 = 1, EP_GELU_BF16 = 2 };

template <int EP>
__global__ __launch_bounds__(256) void gemm_bf16(const uint16_t* __restrict__ A,
                                                 const uint16_t* __restrict__ Bw,
                                                 void* __restrict__ Cout,
                                                 const float* __restrict__ Res,
                                                 int M, int N, int K) {
    __shared__ __align__(16) uint16_t As[128 * 32];   // 8KB row-major [m][k]
    __shared__ __align__(16) uint16_t Bs[128 * 32];   // 8KB transposed [n][k]
    const int tid  = threadIdx.x;
    const int lane = tid & 31;
    const int wave = tid >> 5;
    const int row0 = blockIdx.y * 128;
    const int col0 = blockIdx.x * 128;

    v8f acc[8];
    const v8f vzero = {0.f,0.f,0.f,0.f,0.f,0.f,0.f,0.f};
#pragma unroll
    for (int nt = 0; nt < 8; ++nt) acc[nt] = vzero;

    const int ksteps = K >> 5;
    for (int ks = 0; ks < ksteps; ++ks) {
        const int kk0 = ks << 5;
        // stage A: 128x32 bf16 = 8KB; async b128 = 16B/lane, 2 issues/thread
#pragma unroll
        for (int i = 0; i < 2; ++i) {
            const int e = i * 256 + tid;          // 0..511 16B chunks
            const int r = e >> 2, seg = e & 3;    // 4 chunks per 64B row
            async_load_b128(&As[r * 32 + seg * 8],
                            &A[(size_t)(row0 + r) * K + kk0 + seg * 8]);
        }
        // stage B transposed: 32x128 bf16 = 2048 dwords, 8/thread
#pragma unroll
        for (int i = 0; i < 8; ++i) {
            const int idx = tid * 8 + i;
            const int kk = idx >> 6, nd = idx & 63;
            const uint32_t d =
                *reinterpret_cast<const uint32_t*>(&Bw[(size_t)(kk0 + kk) * N + col0 + nd * 2]);
            Bs[(nd * 2)     * 32 + kk] = (uint16_t)(d & 0xFFFFu);
            Bs[(nd * 2 + 1) * 32 + kk] = (uint16_t)(d >> 16);
        }
        if (ks + 1 < ksteps) {   // global_prefetch_b8 of next K-slab
            __builtin_prefetch(&A[(size_t)(row0 + (tid >> 1)) * K + kk0 + 32]);
            __builtin_prefetch(&Bw[(size_t)(kk0 + 32 + (tid >> 3)) * N + col0]);
        }
        wait_asynccnt0();        // A-tile landed in LDS
        __syncthreads();
        const v16bf af = frag16(&As[(wave * 16) * 32], 32, lane, 0);
#pragma unroll
        for (int nt = 0; nt < 8; ++nt) {
            const v16bf bfg = frag16(&Bs[(nt * 16) * 32], 32, lane, 0);
            acc[nt] = wmma_bf16(af, bfg, acc[nt]);
        }
        __syncthreads();
    }

    const int half = lane >> 4, nl = lane & 15;
#pragma unroll
    for (int nt = 0; nt < 8; ++nt) {
#pragma unroll
        for (int g = 0; g < 8; ++g) {
            const int r = row0 + wave * 16 + g + half * 8;
            const int c = col0 + nt * 16 + nl;
            const float v = acc[nt][g];
            if (EP == EP_RES_F32) {
                ((float*)Cout)[(size_t)r * N + c] = v + Res[(size_t)r * N + c];
            } else if (EP == EP_GELU_BF16) {
                ((uint16_t*)Cout)[(size_t)r * N + c] = f2bf(gelu_exact(v));
            } else {
                ((float*)Cout)[(size_t)r * N + c] = v;
            }
        }
    }
}

// ---------------------------------------------------------------------------
// Fused attention per (b,h): S = QK^T*scale + bias, causal, softmax, O = P@V.
// One block (8 waves, 256 thr) per (b,h); wave w -> 16 query rows.
// ---------------------------------------------------------------------------
__global__ __launch_bounds__(256) void attn_kernel(const float* __restrict__ qkv,
                                                   const float* __restrict__ bias,
                                                   uint16_t* __restrict__ attn_bf) {
    __shared__ uint16_t smem[TT * TT];     // 32KB: [qs | ks] then reused as ps
    __shared__ uint16_t vts[DH * TT];      // 16KB: V transposed [d][t]
    uint16_t* qs  = smem;                  // TT x DH
    uint16_t* ksm = smem + TT * DH;        // TT x DH
    uint16_t* ps  = smem;                  // TT x TT (after phase 1)

    const int bh = blockIdx.x;
    const int b = bh >> 3, h = bh & 7;
    const int tid = threadIdx.x, lane = tid & 31, wave = tid >> 5;

    for (int i = tid; i < TT * DH; i += 256) {
        const int t = i >> 6, d = i & 63;
        const float* rowp = &qkv[(size_t)(b * TT + t) * (3 * DD)];
        qs[t * DH + d]  = f2bf(rowp[h * DH + d]);
        ksm[t * DH + d] = f2bf(rowp[DD + h * DH + d]);
        vts[d * TT + t] = f2bf(rowp[2 * DD + h * DH + d]);
    }
    __syncthreads();

    const int m0 = wave * 16;
    const int half = lane >> 4, nl = lane & 15;
    const v8f vzero = {0.f,0.f,0.f,0.f,0.f,0.f,0.f,0.f};
    const v16bf aq0 = frag16(&qs[m0 * DH], DH, lane, 0);
    const v16bf aq1 = frag16(&qs[m0 * DH], DH, lane, 32);

    float sreg[8][8];
    float rmax[8], rsum[8];
#pragma unroll
    for (int g = 0; g < 8; ++g) { rmax[g] = -3.0e38f; rsum[g] = 0.0f; }
    const float scale = 0.125f;   // 1/sqrt(64)

#pragma unroll
    for (int nt = 0; nt < 8; ++nt) {
        v8f acc = vzero;
        const v16bf b0 = frag16(&ksm[(nt * 16) * DH], DH, lane, 0);
        const v16bf b1 = frag16(&ksm[(nt * 16) * DH], DH, lane, 32);
        acc = wmma_bf16(aq0, b0, acc);
        acc = wmma_bf16(aq1, b1, acc);
#pragma unroll
        for (int g = 0; g < 8; ++g) {
            const int row = m0 + g + half * 8;
            const int col = nt * 16 + nl;
            float v = acc[g] * scale + bias[(size_t)(b * TT + row) * TT + col];
            if (col > row) v = -1e30f;
            sreg[nt][g] = v;
            rmax[g] = fmaxf(rmax[g], v);
        }
    }
#pragma unroll
    for (int g = 0; g < 8; ++g) rmax[g] = xorred_max16(rmax[g]);
#pragma unroll
    for (int nt = 0; nt < 8; ++nt)
#pragma unroll
        for (int g = 0; g < 8; ++g) {
            const float e = __expf(sreg[nt][g] - rmax[g]);
            sreg[nt][g] = e;
            rsum[g] += e;
        }
#pragma unroll
    for (int g = 0; g < 8; ++g) rsum[g] = 1.0f / xorred_add16(rsum[g]);

    __syncthreads();   // qs/ks dead -> reuse as ps
#pragma unroll
    for (int nt = 0; nt < 8; ++nt)
#pragma unroll
        for (int g = 0; g < 8; ++g) {
            const int row = m0 + g + half * 8;
            const int col = nt * 16 + nl;
            ps[row * TT + col] = f2bf(sreg[nt][g] * rsum[g]);
        }
    __syncthreads();

    v16bf ap[4];
#pragma unroll
    for (int k2 = 0; k2 < 4; ++k2) ap[k2] = frag16(&ps[m0 * TT], TT, lane, k2 * 32);
#pragma unroll
    for (int nt = 0; nt < 4; ++nt) {
        v8f acc = vzero;
#pragma unroll
        for (int k2 = 0; k2 < 4; ++k2) {
            const v16bf bv = frag16(&vts[(nt * 16) * TT], TT, lane, k2 * 32);
            acc = wmma_bf16(ap[k2], bv, acc);
        }
#pragma unroll
        for (int g = 0; g < 8; ++g) {
            const int row = b * TT + m0 + g + half * 8;
            const int col = h * DH + nt * 16 + nl;
            attn_bf[(size_t)row * DD + col] = f2bf(acc[g]);
        }
    }
}

// ---------------------------------------------------------------------------
// Fused head: out = LN(GELU(Xrow @ W1), lnw) @ W2
// Block (8 waves) computes 16 rows x all 512 cols; fused GELU+LN+W2 epilogue.
// MODE 0: rows are (b,t), valid if t < num_nodes[b]; MODE 1: edge rows via row_map.
// ---------------------------------------------------------------------------
template <int KDIM, int OUTW, int MODE>
__global__ __launch_bounds__(256) void head_kernel(const uint16_t* __restrict__ Ybf,
                                                   const uint16_t* __restrict__ NEbf,
                                                   const int* __restrict__ tokens,
                                                   const int* __restrict__ row_map,
                                                   const uint16_t* __restrict__ W1bf,
                                                   const float* __restrict__ LNw,
                                                   const float* __restrict__ W2,
                                                   const int* __restrict__ num_nodes,
                                                   const int* __restrict__ ctl,
                                                   float* __restrict__ dout,
                                                   int out_sel) {
    __shared__ uint16_t As[16 * 32];       // 1KB
    __shared__ float    sbuf[16 * DD];     // 32KB, reused: Bs (bf16) then hb (f32)
    uint16_t* Bs = reinterpret_cast<uint16_t*>(sbuf);   // [n][k] 512x32
    float*    hb = sbuf;                                // [16][512]

    const int NN = ctl[0], NE = ctl[1];
    const int r0 = blockIdx.x * 16;
    if (MODE == 1 && r0 >= NE) return;

    const int tid = threadIdx.x, lane = tid & 31, wave = tid >> 5;
    const v8f vzero = {0.f,0.f,0.f,0.f,0.f,0.f,0.f,0.f};
    v8f acc[4];
#pragma unroll
    for (int nt = 0; nt < 4; ++nt) acc[nt] = vzero;

    const int ksteps = KDIM / 32;
    for (int ks = 0; ks < ksteps; ++ks) {
        const int kk0 = ks * 32;
        // stage A (16x32): 256 dwords, 1/thread; edge mode gathers the concat
        {
            const int rr = tid >> 4, cd = tid & 15;
            const int c = kk0 + cd * 2;
            uint32_t val = 0;
            const int r = r0 + rr;
            if (MODE == 0) {
                val = *reinterpret_cast<const uint32_t*>(&Ybf[(size_t)r * DD + c]);
            } else if (r < NE) {
                const int mp = row_map[r];
                const int b = (mp >> 16) & 255, pi = (mp >> 8) & 255, pj = mp & 255;
                const uint16_t* src;
                int cc = c;
                if (c < DD)            { src = &Ybf[(size_t)(b * TT + pi) * DD]; }
                else if (c < 2 * DD)   { src = &Ybf[(size_t)(b * TT + pj) * DD]; cc = c - DD; }
                else {
                    const int tok = tokens[(b * TT + pi + 1) * 2];
                    src = &NEbf[(size_t)tok * DD]; cc = c - 2 * DD;
                }
                val = *reinterpret_cast<const uint32_t*>(src + cc);
            }
            *reinterpret_cast<uint32_t*>(&As[rr * 32 + cd * 2]) = val;
        }
        // stage B transposed (32x512): 8192 dwords, 32/thread
#pragma unroll
        for (int i = 0; i < 32; ++i) {
            const int idx = tid * 32 + i;
            const int kk = idx >> 8, nd = idx & 255;
            const uint32_t d =
                *reinterpret_cast<const uint32_t*>(&W1bf[(size_t)(kk0 + kk) * DD + nd * 2]);
            Bs[(nd * 2)     * 32 + kk] = (uint16_t)(d & 0xFFFFu);
            Bs[(nd * 2 + 1) * 32 + kk] = (uint16_t)(d >> 16);
        }
        __syncthreads();
        const v16bf af = frag16(As, 32, lane, 0);
#pragma unroll
        for (int nt = 0; nt < 4; ++nt) {
            const v16bf bfg = frag16(&Bs[(wave * 64 + nt * 16) * 32], 32, lane, 0);
            acc[nt] = wmma_bf16(af, bfg, acc[nt]);
        }
        __syncthreads();
    }

    // GELU -> hb (Bs is dead now; same storage)
    const int half = lane >> 4, nl = lane & 15;
#pragma unroll
    for (int nt = 0; nt < 4; ++nt)
#pragma unroll
        for (int g = 0; g < 8; ++g) {
            const int rr = g + half * 8;
            const int c = wave * 64 + nt * 16 + nl;
            hb[rr * DD + c] = gelu_exact(acc[nt][g]);
        }
    __syncthreads();

    // LayerNorm + W2: 16 threads per row (tid%16 strides 32 cols)
    const int rr  = tid >> 4;
    const int l16 = tid & 15;
    float hreg[32];
    float s = 0.0f;
#pragma unroll
    for (int j = 0; j < 32; ++j) {
        hreg[j] = hb[rr * DD + j * 16 + l16];
        s += hreg[j];
    }
    s = xorred_add16(s);
    const float mu = s * (1.0f / (float)DD);
    float v = 0.0f;
#pragma unroll
    for (int j = 0; j < 32; ++j) { const float d = hreg[j] - mu; v += d * d; }
    v = xorred_add16(v);
    const float rstd = rsqrtf(v * (1.0f / (float)DD) + 1e-5f);
#pragma unroll
    for (int j = 0; j < 32; ++j)
        hreg[j] = (hreg[j] - mu) * rstd * LNw[j * 16 + l16];

    float outv[OUTW];
#pragma unroll
    for (int jw = 0; jw < OUTW; ++jw) {
        float p = 0.0f;
#pragma unroll
        for (int j = 0; j < 32; ++j)
            p += hreg[j] * W2[(j * 16 + l16) * OUTW + jw];
        outv[jw] = xorred_add16(p);
    }

    if (l16 == 0) {
        const int r = r0 + rr;
        long orow = -1, obase = 0;
        if (MODE == 1) {
            if (r < NE) orow = r;
        } else {
            const int b = r >> 7, t = r & 127;
            if (t < num_nodes[b]) orow = (long)ctl[2 + b] + t;
            obase = (out_sel == 1) ? (long)NE * 5 : (long)NE * 5 + (long)NN * 16;
        }
        if (orow >= 0) {
#pragma unroll
            for (int jw = 0; jw < OUTW; ++jw)
                dout[obase + orow * OUTW + jw] = outv[jw];
        }
    }
}

// ---------------------------------------------------------------------------
// Host-side orchestration
// ---------------------------------------------------------------------------
extern "C" void kernel_launch(void* const* d_in, const int* in_sizes, int n_in,
                              void* d_out, int out_size, void* d_ws, size_t ws_size,
                              hipStream_t stream) {
    const int*   tokens    = (const int*)d_in[0];
    const int*   num_nodes = (const int*)d_in[1];
    const float* bias      = (const float*)d_in[2];
    const float* node_emb  = (const float*)d_in[3];
    const float* Wqkv      = (const float*)d_in[4];
    const float* Wo        = (const float*)d_in[5];
    const float* ln1       = (const float*)d_in[6];
    const float* ln2       = (const float*)d_in[7];
    const float* W1        = (const float*)d_in[8];
    const float* W2        = (const float*)d_in[9];
    const float* lnf       = (const float*)d_in[10];
    const float* node_w1   = (const float*)d_in[11];
    const float* node_ln   = (const float*)d_in[12];
    const float* node_w2   = (const float*)d_in[13];
    const float* spec_w1   = (const float*)d_in[14];
    const float* spec_ln   = (const float*)d_in[15];
    const float* spec_w2   = (const float*)d_in[16];
    const float* edge_w1   = (const float*)d_in[17];
    const float* edge_ln   = (const float*)d_in[18];
    const float* edge_w2   = (const float*)d_in[19];
    (void)in_sizes; (void)n_in; (void)out_size; (void)ws_size;

    // --- workspace carving ---
    char* p = (char*)d_ws;
    auto carve = [&](size_t bytes) {
        char* r = p;
        p += (bytes + 255) & ~(size_t)255;
        return (void*)r;
    };
    int*      ctl      = (int*)     carve(512);
    int*      row_map  = (int*)     carve((size_t)NE_MAX * 4);
    float*    xbuf     = (float*)   carve((size_t)BT * DD * 4);
    float*    x2buf    = (float*)   carve((size_t)BT * DD * 4);
    float*    qkvbuf   = (float*)   carve((size_t)BT * 3 * DD * 4);
    uint16_t* hbf      = (uint16_t*)carve((size_t)BT * DD * 2);
    uint16_t* attnbf   = (uint16_t*)carve((size_t)BT * DD * 2);
    uint16_t* gbuf     = (uint16_t*)carve((size_t)BT * DFF * 2);
    uint16_t* ybf      = (uint16_t*)carve((size_t)BT * DD * 2);
    uint16_t* Wqkv_bf  = (uint16_t*)carve((size_t)LL * DD * 3 * DD * 2);
    uint16_t* Wo_bf    = (uint16_t*)carve((size_t)LL * DD * DD * 2);
    uint16_t* W1_bf    = (uint16_t*)carve((size_t)LL * DD * DFF * 2);
    uint16_t* W2_bf    = (uint16_t*)carve((size_t)LL * DFF * DD * 2);
    uint16_t* nodew1bf = (uint16_t*)carve((size_t)DD * DD * 2);
    uint16_t* specw1bf = (uint16_t*)carve((size_t)DD * DD * 2);
    uint16_t* edgew1bf = (uint16_t*)carve((size_t)3 * DD * DD * 2);
    uint16_t* nebf     = (uint16_t*)carve((size_t)32 * DD * 2);

    // --- index prep ---
    prep_kernel<<<1, 32, 0, stream>>>(num_nodes, ctl);
    rowmap_kernel<<<BB * (TT - 1), 128, 0, stream>>>(num_nodes, ctl, row_map);

    // --- weight conversion to bf16 ---
    auto conv = [&](const float* s, uint16_t* d, size_t n) {
        convert_bf16_kernel<<<(unsigned)((n + 255) / 256), 256, 0, stream>>>(s, d, (int)n);
    };
    conv(Wqkv,     Wqkv_bf,  (size_t)LL * DD * 3 * DD);
    conv(Wo,       Wo_bf,    (size_t)LL * DD * DD);
    conv(W1,       W1_bf,    (size_t)LL * DD * DFF);
    conv(W2,       W2_bf,    (size_t)LL * DFF * DD);
    conv(node_w1,  nodew1bf, (size_t)DD * DD);
    conv(spec_w1,  specw1bf, (size_t)DD * DD);
    conv(edge_w1,  edgew1bf, (size_t)3 * DD * DD);
    conv(node_emb, nebf,     (size_t)32 * DD);

    // --- embedding ---
    embed_kernel<<<BT, 128, 0, stream>>>(tokens, node_emb, xbuf);

    // --- transformer layers ---
    for (int l = 0; l < LL; ++l) {
        ln_bf16_kernel<<<BT, 256, 0, stream>>>(xbuf, ln1 + (size_t)l * DD, hbf);
        gemm_bf16<EP_F32><<<dim3(3 * DD / 128, BT / 128), 256, 0, stream>>>(
            hbf, Wqkv_bf + (size_t)l * DD * 3 * DD, qkvbuf, nullptr, BT, 3 * DD, DD);
        attn_kernel<<<BB * HH, 256, 0, stream>>>(qkvbuf, bias, attnbf);
        gemm_bf16<EP_RES_F32><<<dim3(DD / 128, BT / 128), 256, 0, stream>>>(
            attnbf, Wo_bf + (size_t)l * DD * DD, x2buf, xbuf, BT, DD, DD);
        ln_bf16_kernel<<<BT, 256, 0, stream>>>(x2buf, ln2 + (size_t)l * DD, hbf);
        gemm_bf16<EP_GELU_BF16><<<dim3(DFF / 128, BT / 128), 256, 0, stream>>>(
            hbf, W1_bf + (size_t)l * DD * DFF, gbuf, nullptr, BT, DFF, DD);
        gemm_bf16<EP_RES_F32><<<dim3(DD / 128, BT / 128), 256, 0, stream>>>(
            gbuf, W2_bf + (size_t)l * DFF * DD, xbuf, x2buf, BT, DD, DFF);
    }

    // --- final LN ---
    ln_bf16_kernel<<<BT, 256, 0, stream>>>(xbuf, lnf, ybf);

    // --- heads (fused GEMM + GELU + LN + W2) ---
    head_kernel<DD, 16, 0><<<BT / 16, 256, 0, stream>>>(
        ybf, nullptr, nullptr, nullptr, nodew1bf, node_ln, node_w2,
        num_nodes, ctl, (float*)d_out, 1);
    head_kernel<DD, 4, 0><<<BT / 16, 256, 0, stream>>>(
        ybf, nullptr, nullptr, nullptr, specw1bf, spec_ln, spec_w2,
        num_nodes, ctl, (float*)d_out, 2);
    head_kernel<3 * DD, 5, 1><<<NE_MAX / 16, 256, 0, stream>>>(
        ybf, nebf, tokens, row_map, edgew1bf, edge_ln, edge_w2,
        num_nodes, ctl, (float*)d_out, 0);
}
